// SelfAttention_19104014533046
// MI455X (gfx1250) — compile-verified, matrix operand order checked by reading, not run on hardware
//
#include <hip/hip_runtime.h>
#include <hip/hip_bf16.h>

typedef __attribute__((ext_vector_type(16))) _Float16 v16h;
typedef __attribute__((ext_vector_type(8)))  _Float16 v8h;
typedef __attribute__((ext_vector_type(8)))  float    v8f;

#define Bq 4
#define Cc 256
#define Nn 4096   // 64*64 pooled spatial
#define Dd 32     // C/8

// Build a 16-bit A/B WMMA fragment from two contiguous 16-byte chunks.
// A-layout (16x32 f16): lanes 0-15 hold K {0..7,16..23}, lanes 16-31 K {8..15,24..31};
// we use the mirrored pattern for B (per-lane column, K contiguous in memory).
__device__ __forceinline__ v16h frag_from(const _Float16* base, int off0, int off1) {
    v8h a = *(const v8h*)(base + off0);
    v8h b = *(const v8h*)(base + off1);
    v16h r;
#pragma unroll
    for (int i = 0; i < 8; ++i) { r[i] = a[i]; r[i + 8] = b[i]; }
    return r;
}

// ---------------- float -> f16 weight conversion ----------------
__global__ __launch_bounds__(256) void f2h_kernel(const float* __restrict__ src,
                                                  _Float16* __restrict__ dst, int n) {
    int i = blockIdx.x * 256 + threadIdx.x;
    if (i < n) dst[i] = (_Float16)src[i];
}

// ---------------- 4x4 average pool, f32 -> f16 [B,C,N] ----------------
__global__ __launch_bounds__(256) void pool_kernel(const float* __restrict__ x,
                                                   _Float16* __restrict__ xh) {
    int gid = blockIdx.x * 256 + threadIdx.x;          // 4M threads
    int wp = gid & 63, hp = (gid >> 6) & 63, c = (gid >> 12) & 255, b = gid >> 20;
    const float* base = x + (((size_t)(b * Cc + c) * 256 + hp * 4) * 256) + wp * 4;
    float s = 0.f;
#pragma unroll
    for (int j = 0; j < 4; ++j) {
        float4 v = *(const float4*)(base + j * 256);
        s += v.x + v.y + v.z + v.w;
    }
    xh[(size_t)(b * Cc + c) * Nn + hp * 64 + wp] = (_Float16)(s * 0.0625f);
}

// ---------------- transpose [B,C,N] -> [B,N,C] ----------------
__global__ __launch_bounds__(256) void transpose_kernel(const _Float16* __restrict__ xh,
                                                        _Float16* __restrict__ xhT) {
    __shared__ _Float16 tile[32][33];
    int bid = blockIdx.x;                               // 4*8*128 blocks
    int ntile = bid & 127, ctile = (bid >> 7) & 7, b = bid >> 10;
    int c0 = ctile * 32, n0 = ntile * 32;
    int t = threadIdx.x;
#pragma unroll
    for (int i = 0; i < 4; ++i) {
        int idx = t + i * 256;
        int r = idx >> 5, col = idx & 31;
        tile[r][col] = xh[(size_t)(b * Cc + c0 + r) * Nn + n0 + col];
    }
    __syncthreads();
#pragma unroll
    for (int i = 0; i < 4; ++i) {
        int idx = t + i * 256;
        int r = idx >> 5, col = idx & 31;               // r: n offset, col: c offset
        xhT[(size_t)(b * Nn + n0 + r) * Cc + c0 + col] = tile[col][r];
    }
}

// ---------------- q/k GEMM: outT[b,n,d] = sum_c xhT[b,n,c]*W[d,c] + bias[d] ----------------
__global__ __launch_bounds__(256) void qk_kernel(const _Float16* __restrict__ Wh,
                                                 const float* __restrict__ bias,
                                                 const _Float16* __restrict__ xhT,
                                                 _Float16* __restrict__ outT) {
    const int t = threadIdx.x;
    const int lane = t & 31, half = lane >> 4, l = lane & 15;
    const int g = blockIdx.x * 8 + (t >> 5);            // 2048 tile-waves
    const int dt = g & 1, nt = (g >> 1) & 255, b = g >> 9;
    const int nbase = nt * 16, dbase = dt * 16;
    const _Float16* arow = xhT + ((size_t)(b * Nn + nbase + l)) * Cc;  // rows n, K=c
    const _Float16* brow = Wh + (size_t)(dbase + l) * Cc;              // col d,  K=c
    v8f acc = {};
#pragma unroll
    for (int kk = 0; kk < 8; ++kk) {
        int k0 = kk * 32 + (half ? 8 : 0), k1 = kk * 32 + (half ? 24 : 16);
        v16h af = frag_from(arow, k0, k1);
        v16h bf = frag_from(brow, k0, k1);
        acc = __builtin_amdgcn_wmma_f32_16x16x32_f16(false, af, false, bf, (short)0, acc, false, false);
    }
    const float bval = bias[dbase + l];
#pragma unroll
    for (int r = 0; r < 8; ++r) {
        int n = nbase + r + half * 8;                   // D-layout rows
        outT[(size_t)(b * Nn + n) * Dd + dbase + l] = (_Float16)(acc[r] + bval);
    }
}

// ---------------- v GEMM: vh[b,e,m] = sum_c Wv[e,c]*xhT[b,m,c] + bv[e] ----------------
__global__ __launch_bounds__(256) void vgemm_kernel(const _Float16* __restrict__ Wvh,
                                                    const float* __restrict__ bv,
                                                    const _Float16* __restrict__ xhT,
                                                    _Float16* __restrict__ vh) {
    const int t = threadIdx.x;
    const int lane = t & 31, half = lane >> 4, l = lane & 15;
    const int g = blockIdx.x * 8 + (t >> 5);            // 16384 tile-waves
    const int mt = g & 255, et = (g >> 8) & 15, b = g >> 12;
    const int ebase = et * 16, mbase = mt * 16;
    const _Float16* arow = Wvh + (size_t)(ebase + l) * Cc;             // rows e, K=c
    const _Float16* brow = xhT + ((size_t)(b * Nn + mbase + l)) * Cc;  // col m,  K=c
    v8f acc = {};
#pragma unroll
    for (int kk = 0; kk < 8; ++kk) {
        int k0 = kk * 32 + (half ? 8 : 0), k1 = kk * 32 + (half ? 24 : 16);
        v16h af = frag_from(arow, k0, k1);
        v16h bf = frag_from(brow, k0, k1);
        acc = __builtin_amdgcn_wmma_f32_16x16x32_f16(false, af, false, bf, (short)0, acc, false, false);
    }
#pragma unroll
    for (int r = 0; r < 8; ++r) {
        int e = ebase + r + half * 8;
        vh[(size_t)(b * Cc + e) * Nn + mbase + l] = (_Float16)(acc[r] + bv[e]);
    }
}

// ---------------- fused attention (flash style), per block: (b, 16 query rows) ----------------
__global__ __launch_bounds__(256) void attn_kernel(const _Float16* __restrict__ qT,
                                                   const _Float16* __restrict__ kT,
                                                   const _Float16* __restrict__ vh,
                                                   float* __restrict__ op) {
    const int t = threadIdx.x;
    const int wave = t >> 5, lane = t & 31, half = lane >> 4, l = lane & 15;
    const int nt = blockIdx.x & 255, b = blockIdx.x >> 8;
    const int nbase = nt * 16;

    __shared__ __align__(16) float    s_raw[16 * 260];   // raw scores, padded stride
    __shared__ __align__(16) _Float16 p_lds[16 * 264];   // exp(P) f16, padded stride
    __shared__ float pmax[256], psum[256];
    __shared__ float rowmax[16], rowsum[16], alpha_s[16];

    if (t < 16) { rowmax[t] = -1e30f; rowsum[t] = 0.f; }
    __syncthreads();

    // q A-fragment: rows n = nbase+l, K = d (exactly 32 -> one WMMA per tile)
    const _Float16* qrow = qT + ((size_t)(b * Nn + nbase + l)) * Dd;
    const v16h qfrag = frag_from(qrow, half ? 8 : 0, half ? 24 : 16);

    v8f acc0 = {}, acc1 = {};
    const int ct0 = wave * 2, ct1 = wave * 2 + 1;

    for (int ch = 0; ch < 16; ++ch) {
        const int mchunk = ch * 256;

        // ---- S tiles: wave w computes m-tiles 2w, 2w+1
#pragma unroll
        for (int mt2 = 0; mt2 < 2; ++mt2) {
            const int mt = wave * 2 + mt2;
            const int m = mchunk + mt * 16 + l;
            const _Float16* krow = kT + ((size_t)(b * Nn + m)) * Dd;
            v16h kfrag = frag_from(krow, half ? 8 : 0, half ? 24 : 16);
            v8f s = {};
            s = __builtin_amdgcn_wmma_f32_16x16x32_f16(false, qfrag, false, kfrag, (short)0, s, false, false);
#pragma unroll
            for (int r = 0; r < 8; ++r)
                s_raw[(r + half * 8) * 260 + mt * 16 + l] = s[r];
        }
        __syncthreads();

        // ---- per-segment max
        {
            const int r = t >> 4, seg = t & 15;
            float mx = -1e30f;
#pragma unroll
            for (int j = 0; j < 16; ++j) mx = fmaxf(mx, s_raw[r * 260 + seg * 16 + j]);
            pmax[t] = mx;
        }
        __syncthreads();
        if (t < 16) {
            float mnew = rowmax[t];
#pragma unroll
            for (int sg = 0; sg < 16; ++sg) mnew = fmaxf(mnew, pmax[t * 16 + sg]);
            float a = __expf(rowmax[t] - mnew);
            rowmax[t] = mnew; alpha_s[t] = a; rowsum[t] *= a;
        }
        __syncthreads();
        // ---- exp + partial sums, stage P as f16
        {
            const int r = t >> 4, seg = t & 15;
            const float mnew = rowmax[r];
            float sloc = 0.f;
#pragma unroll
            for (int j = 0; j < 16; ++j) {
                float e = __expf(s_raw[r * 260 + seg * 16 + j] - mnew);
                p_lds[r * 264 + seg * 16 + j] = (_Float16)e;
                sloc += e;
            }
            psum[t] = sloc;
        }
        __syncthreads();
        if (t < 16) {
            float s2 = 0.f;
#pragma unroll
            for (int sg = 0; sg < 16; ++sg) s2 += psum[t * 16 + sg];
            rowsum[t] += s2;
        }

        // ---- O accumulation: wave w owns C-tiles 2w, 2w+1; rescale then 8 WMMAs each
        const float a = alpha_s[l];
#pragma unroll
        for (int r = 0; r < 8; ++r) { acc0[r] *= a; acc1[r] *= a; }

        const _Float16* vrow0 = vh + ((size_t)(b * Cc + ct0 * 16 + l)) * Nn + mchunk;
        const _Float16* vrow1 = vh + ((size_t)(b * Cc + ct1 * 16 + l)) * Nn + mchunk;
        const _Float16* prow  = p_lds + l * 264;
#pragma unroll
        for (int kk = 0; kk < 8; ++kk) {
            const int k0 = kk * 32 + (half ? 8 : 0);
            const int k1 = kk * 32 + (half ? 24 : 16);
            v16h pf = frag_from(prow, k0, k1);
            v16h a0 = frag_from(vrow0, k0, k1);
            acc0 = __builtin_amdgcn_wmma_f32_16x16x32_f16(false, a0, false, pf, (short)0, acc0, false, false);
            v16h a1 = frag_from(vrow1, k0, k1);
            acc1 = __builtin_amdgcn_wmma_f32_16x16x32_f16(false, a1, false, pf, (short)0, acc1, false, false);
        }
        __syncthreads();   // protect s_raw/p_lds/pmax/psum reuse next chunk
    }

    const float inv = 1.f / rowsum[l];
    const int n = nbase + l;
#pragma unroll
    for (int r = 0; r < 8; ++r) {
        int c0 = ct0 * 16 + r + half * 8;
        int c1 = ct1 * 16 + r + half * 8;
        op[(size_t)(b * Cc + c0) * Nn + n] = acc0[r] * inv;
        op[(size_t)(b * Cc + c1) * Nn + n] = acc1[r] * inv;
    }
}

// ---------------- bilinear 4x upsample (half-pixel) + residual ----------------
__global__ __launch_bounds__(256) void upsample_kernel(const float* __restrict__ op,
                                                       const float* __restrict__ x,
                                                       const float* __restrict__ gamma,
                                                       float* __restrict__ out) {
    int gid = blockIdx.x * 256 + threadIdx.x;           // 16.7M threads, 4 px each
    int w4 = gid & 63, h = (gid >> 6) & 255, c = (gid >> 14) & 255, b = gid >> 22;
    const float g = gamma[0];

    float fh = (h + 0.5f) * 0.25f - 0.5f;
    float fh0 = floorf(fh);
    float wh = fh - fh0;
    int ih0 = (int)fh0, ih1 = ih0 + 1;
    ih0 = ih0 < 0 ? 0 : (ih0 > 63 ? 63 : ih0);
    ih1 = ih1 < 0 ? 0 : (ih1 > 63 ? 63 : ih1);

    const float* prow0 = op + (size_t)(b * Cc + c) * Nn + ih0 * 64;
    const float* prow1 = op + (size_t)(b * Cc + c) * Nn + ih1 * 64;
    size_t xoff = ((size_t)(b * Cc + c) * 256 + h) * 256 + w4 * 4;
    float4 xv = *(const float4*)(x + xoff);

    float res[4];
#pragma unroll
    for (int j = 0; j < 4; ++j) {
        int w = w4 * 4 + j;
        float fw = (w + 0.5f) * 0.25f - 0.5f;
        float fw0 = floorf(fw);
        float ww = fw - fw0;
        int iw0 = (int)fw0, iw1 = iw0 + 1;
        iw0 = iw0 < 0 ? 0 : (iw0 > 63 ? 63 : iw0);
        iw1 = iw1 < 0 ? 0 : (iw1 > 63 ? 63 : iw1);
        float v00 = prow0[iw0], v01 = prow0[iw1];
        float v10 = prow1[iw0], v11 = prow1[iw1];
        float top = v00 + ww * (v01 - v00);
        float bot = v10 + ww * (v11 - v10);
        res[j] = top + wh * (bot - top);
    }
    float4 o;
    o.x = g * res[0] + xv.x; o.y = g * res[1] + xv.y;
    o.z = g * res[2] + xv.z; o.w = g * res[3] + xv.w;
    *(float4*)(out + xoff) = o;
}

extern "C" void kernel_launch(void* const* d_in, const int* in_sizes, int n_in,
                              void* d_out, int out_size, void* d_ws, size_t ws_size,
                              hipStream_t stream) {
    const float* x     = (const float*)d_in[0];
    const float* Wq    = (const float*)d_in[1];
    const float* bq    = (const float*)d_in[2];
    const float* Wk    = (const float*)d_in[3];
    const float* bk    = (const float*)d_in[4];
    const float* Wv    = (const float*)d_in[5];
    const float* bv    = (const float*)d_in[6];
    const float* gamma = (const float*)d_in[7];
    float* out = (float*)d_out;

    char* ws = (char*)d_ws;
    _Float16* xh  = (_Float16*)ws; ws += (size_t)Bq * Cc * Nn * 2;   // 8 MB
    _Float16* xhT = (_Float16*)ws; ws += (size_t)Bq * Nn * Cc * 2;   // 8 MB
    _Float16* qT  = (_Float16*)ws; ws += (size_t)Bq * Nn * Dd * 2;   // 1 MB
    _Float16* kT  = (_Float16*)ws; ws += (size_t)Bq * Nn * Dd * 2;   // 1 MB
    _Float16* vh  = (_Float16*)ws; ws += (size_t)Bq * Cc * Nn * 2;   // 8 MB
    float*    op  = (float*)ws;    ws += (size_t)Bq * Cc * Nn * 4;   // 16 MB
    _Float16* Wqh = (_Float16*)ws; ws += (size_t)Dd * Cc * 2;
    _Float16* Wkh = (_Float16*)ws; ws += (size_t)Dd * Cc * 2;
    _Float16* Wvh = (_Float16*)ws; ws += (size_t)Cc * Cc * 2;

    f2h_kernel<<<(Dd * Cc + 255) / 256, 256, 0, stream>>>(Wq, Wqh, Dd * Cc);
    f2h_kernel<<<(Dd * Cc + 255) / 256, 256, 0, stream>>>(Wk, Wkh, Dd * Cc);
    f2h_kernel<<<(Cc * Cc + 255) / 256, 256, 0, stream>>>(Wv, Wvh, Cc * Cc);

    pool_kernel<<<16384, 256, 0, stream>>>(x, xh);
    transpose_kernel<<<4096, 256, 0, stream>>>(xh, xhT);
    qk_kernel<<<256, 256, 0, stream>>>(Wqh, bq, xhT, qT);
    qk_kernel<<<256, 256, 0, stream>>>(Wkh, bk, xhT, kT);
    vgemm_kernel<<<2048, 256, 0, stream>>>(Wvh, bv, xhT, vh);
    attn_kernel<<<1024, 256, 0, stream>>>(qT, kT, vh, op);
    upsample_kernel<<<65536, 256, 0, stream>>>(op, x, gamma, out);
}
